// MaskRelationMultiHeadAttention_223338299862
// MI455X (gfx1250) — matrix operand order repaired and checked by reading, not compiled, not run on hardware
//
#include <hip/hip_runtime.h>
#include <hip/hip_bf16.h>
#include <math.h>

#define BB 16
#define SS 4096
#define HH 16
#define DK 64
#define DM 1024   // d_model == QD == KVD

typedef __bf16 bf16_t;
typedef __attribute__((ext_vector_type(16))) __bf16 v16bf;
typedef __attribute__((ext_vector_type(8)))  __bf16 v8bf;
typedef __attribute__((ext_vector_type(8)))  float  v8f;
typedef __attribute__((ext_vector_type(4)))  float  v4f;

// ---------------------------------------------------------------------------
// Pass 1a: qh[b,m] = q[b,:] @ Wq[:,m] + bq[m]    (16 x 1024, tiny)
// ---------------------------------------------------------------------------
__global__ __launch_bounds__(256) void qh_kernel(const float* __restrict__ q,
                                                 const float* __restrict__ Wq,
                                                 const float* __restrict__ bq,
                                                 float* __restrict__ qh) {
  const int b = blockIdx.x;
  const float* qrow = q + (size_t)b * DM;
#pragma unroll
  for (int i = 0; i < 4; ++i) {
    const int m = threadIdx.x + i * 256;
    float acc = bq[m];
    for (int d = 0; d < DM; ++d)
      acc = fmaf(qrow[d], Wq[(size_t)d * DM + m], acc);
    qh[(size_t)b * DM + m] = acc;
  }
}

// ---------------------------------------------------------------------------
// Pass 1b: qt[b,h,d] = sum_j qh[b,h*64+j] * Wk[d, h*64+j]  (stored bf16)
//          sb[b,h]   = sum_j qh[b,h*64+j] * bk[h*64+j]     (score bias)
// ---------------------------------------------------------------------------
__global__ __launch_bounds__(256) void qt_kernel(const float* __restrict__ qh,
                                                 const float* __restrict__ Wk,
                                                 const float* __restrict__ bk,
                                                 bf16_t* __restrict__ qt,
                                                 float* __restrict__ sb) {
  const int h = blockIdx.x, b = blockIdx.y;
  __shared__ float qs[DK];
  if (threadIdx.x < DK) qs[threadIdx.x] = qh[(size_t)b * DM + h * DK + threadIdx.x];
  __syncthreads();
  if (threadIdx.x == 0) {
    float a = 0.f;
    for (int j = 0; j < DK; ++j) a = fmaf(qs[j], bk[h * DK + j], a);
    sb[b * HH + h] = a;
  }
#pragma unroll
  for (int i = 0; i < 4; ++i) {
    const int d = threadIdx.x + i * 256;
    const float* wr = Wk + (size_t)d * DM + h * DK;
    float acc = 0.f;
    for (int j = 0; j < DK; ++j) acc = fmaf(qs[j], wr[j], acc);
    qt[((size_t)(b * HH + h)) * DM + d] = (bf16_t)acc;
  }
}

// ---------------------------------------------------------------------------
// Pass 2: scores[b,h,s] = (qt[b,h,:] . k[b,s,:] + sb[b,h]) / 8
// One wave per 16-token tile: WMMA A = qt (16 heads x K), B = k-tile^T.
// Streams all of k exactly once (256 MB).
// ---------------------------------------------------------------------------
__global__ __launch_bounds__(256) void score_kernel(const float* __restrict__ k,
                                                    const bf16_t* __restrict__ qt,
                                                    const float* __restrict__ sb,
                                                    float* __restrict__ scores) {
  const int lane = threadIdx.x & 31;
  const int wave = threadIdx.x >> 5;
  const int b    = blockIdx.y;
  const int s0   = (blockIdx.x * 8 + wave) * 16;
  const int m    = lane & 15;   // A row (head) and B column (token)
  const int half = lane >> 4;

  const float*  krow  = k  + ((size_t)b * SS + (s0 + m)) * DM + half * 16;
  const bf16_t* qbase = qt + ((size_t)(b * HH + m)) * DM + half * 16;

  v8f c = {};
  for (int kk = 0; kk < DM; kk += 32) {
    __builtin_prefetch(krow + kk + 512, 0, 0);          // global_prefetch_b8
    // A fragment: this code's contiguous k order (K 0-15 for lanes 0-15; 16-31 for hi half)
    v8bf a0 = *(const v8bf*)(qbase + kk);
    v8bf a1 = *(const v8bf*)(qbase + kk + 8);
    v16bf a = __builtin_shufflevector(a0, a1, 0,1,2,3,4,5,6,7,8,9,10,11,12,13,14,15);
    // B fragment: lane holds 16 consecutive k values of its token row
    const v4f* kp = (const v4f*)(krow + kk);
    v16bf bb;
#pragma unroll
    for (int q4 = 0; q4 < 4; ++q4) {
      v4f f = kp[q4];
      bb[q4 * 4 + 0] = (bf16_t)f[0];
      bb[q4 * 4 + 1] = (bf16_t)f[1];
      bb[q4 * 4 + 2] = (bf16_t)f[2];
      bb[q4 * 4 + 3] = (bf16_t)f[3];
    }
    c = __builtin_amdgcn_wmma_f32_16x16x32_bf16(false, a, false, bb,
                                                (short)0, c, false, false);
  }
#pragma unroll
  for (int j = 0; j < 8; ++j) {
    const int head = j + 8 * half;
    scores[((size_t)(b * HH + head)) * SS + s0 + m] =
        (c[j] + sb[b * HH + head]) * 0.125f;
  }
}

// ---------------------------------------------------------------------------
// Pass 3: per (b,h) dual softmax -> bf16 weights.
// plain: softmax(scores). masked: softmax(where(mask, 0, scores)).
// ---------------------------------------------------------------------------
__device__ __forceinline__ float blockMax(float v, float* red) {
  const int t = threadIdx.x;
  red[t] = v; __syncthreads();
  for (int o = 128; o > 0; o >>= 1) { if (t < o) red[t] = fmaxf(red[t], red[t + o]); __syncthreads(); }
  float r = red[0]; __syncthreads(); return r;
}
__device__ __forceinline__ float blockSum(float v, float* red) {
  const int t = threadIdx.x;
  red[t] = v; __syncthreads();
  for (int o = 128; o > 0; o >>= 1) { if (t < o) red[t] += red[t + o]; __syncthreads(); }
  float r = red[0]; __syncthreads(); return r;
}

__global__ __launch_bounds__(256) void softmax_kernel(const float* __restrict__ scores,
                                                      const unsigned char* __restrict__ mask,
                                                      bf16_t* __restrict__ wplain,
                                                      bf16_t* __restrict__ wmask) {
  const int bh = blockIdx.x;                       // b*16 + h
  const float* sc = scores + (size_t)bh * SS;
  const unsigned char* mk = mask + (size_t)bh * SS;
  __shared__ float red[256];

  float vals[16]; unsigned char mv[16];
  float mx = -3.0e38f, mxm = -3.0e38f;
#pragma unroll
  for (int i = 0; i < 16; ++i) {
    const int s = i * 256 + threadIdx.x;
    vals[i] = sc[s]; mv[i] = mk[s];
    mx  = fmaxf(mx, vals[i]);
    mxm = fmaxf(mxm, mv[i] ? 0.f : vals[i]);
  }
  mx  = blockMax(mx, red);
  mxm = blockMax(mxm, red);
  float sum = 0.f, summ = 0.f;
#pragma unroll
  for (int i = 0; i < 16; ++i) {
    sum  += __expf(vals[i] - mx);
    summ += __expf((mv[i] ? 0.f : vals[i]) - mxm);
  }
  const float Z  = blockSum(sum, red);
  const float Zm = blockSum(summ, red);
  const float rZ = 1.f / Z, rZm = 1.f / Zm;
#pragma unroll
  for (int i = 0; i < 16; ++i) {
    const int s = i * 256 + threadIdx.x;
    wplain[(size_t)bh * SS + s] = (bf16_t)(__expf(vals[i] - mx) * rZ);
    wmask [(size_t)bh * SS + s] = (bf16_t)(__expf((mv[i] ? 0.f : vals[i]) - mxm) * rZm);
  }
}

// ---------------------------------------------------------------------------
// Pass 4: vbar[path,b,h,d] = sum_s w[path,b,h,s] * v[b,s,d]
// Per wave: one 16-col tile of d, both paths (two accumulators share B frag).
// Streams all of v exactly once (256 MB).
// ---------------------------------------------------------------------------
__global__ __launch_bounds__(256) void vbar_kernel(const float* __restrict__ v,
                                                   const bf16_t* __restrict__ wp,
                                                   const bf16_t* __restrict__ wm,
                                                   float* __restrict__ vbar) {
  const int lane = threadIdx.x & 31;
  const int wave = threadIdx.x >> 5;
  const int b    = blockIdx.y;
  const int n0   = (blockIdx.x * 8 + wave) * 16;   // d-tile base
  const int m    = lane & 15;
  const int half = lane >> 4;

  const bf16_t* wpb  = wp + ((size_t)(b * HH + m)) * SS + half * 16;
  const bf16_t* wmb  = wm + ((size_t)(b * HH + m)) * SS + half * 16;
  const float*  vcol = v + (size_t)b * SS * DM + n0 + m;

  v8f c0 = {}, c1 = {};
  for (int kk = 0; kk < SS; kk += 32) {
    // A fragments: plain & masked weights (bf16, contiguous in s)
    v8bf p0 = *(const v8bf*)(wpb + kk);
    v8bf p1 = *(const v8bf*)(wpb + kk + 8);
    v16bf a0 = __builtin_shufflevector(p0, p1, 0,1,2,3,4,5,6,7,8,9,10,11,12,13,14,15);
    v8bf q0 = *(const v8bf*)(wmb + kk);
    v8bf q1 = *(const v8bf*)(wmb + kk + 8);
    v16bf a1 = __builtin_shufflevector(q0, q1, 0,1,2,3,4,5,6,7,8,9,10,11,12,13,14,15);
    // B fragment: column n0+m of v, K rows kk+half*16 .. +15 (64B coalesced per half-wave)
    const size_t rbase = (size_t)(kk + half * 16) * DM;
    __builtin_prefetch(vcol + rbase + (size_t)32 * DM, 0, 0);
    v16bf bb;
#pragma unroll
    for (int j = 0; j < 16; ++j)
      bb[j] = (bf16_t)vcol[rbase + (size_t)j * DM];
    c0 = __builtin_amdgcn_wmma_f32_16x16x32_bf16(false, a0, false, bb,
                                                 (short)0, c0, false, false);
    c1 = __builtin_amdgcn_wmma_f32_16x16x32_bf16(false, a1, false, bb,
                                                 (short)0, c1, false, false);
  }
#pragma unroll
  for (int j = 0; j < 8; ++j) {
    const int head = j + 8 * half;
    vbar[((size_t)((0 * BB + b) * HH + head)) * DM + n0 + m] = c0[j];
    vbar[((size_t)((1 * BB + b) * HH + head)) * DM + n0 + m] = c1[j];
  }
}

// ---------------------------------------------------------------------------
// Pass 5: U[pb,m] = vbar[pb, m/64, :] . Wv[:, m] + bv[m]   (pb = path*16+b)
// ---------------------------------------------------------------------------
__global__ __launch_bounds__(256) void uproj_kernel(const float* __restrict__ vbar,
                                                    const float* __restrict__ Wv,
                                                    const float* __restrict__ bv,
                                                    float* __restrict__ U) {
  const int pb = blockIdx.x;
#pragma unroll
  for (int i = 0; i < 4; ++i) {
    const int m = threadIdx.x + i * 256;
    const float* vb = vbar + ((size_t)pb * HH + (m >> 6)) * DM;
    float acc = bv[m];
    for (int d = 0; d < DM; ++d)
      acc = fmaf(vb[d], Wv[(size_t)d * DM + m], acc);
    U[(size_t)pb * DM + m] = acc;
  }
}

// ---------------------------------------------------------------------------
// Pass 6: out[pb,o] = U[pb,:] @ Wo[:,o] + bo[o]  -> d_out (plain ‖ masked)
// ---------------------------------------------------------------------------
__global__ __launch_bounds__(256) void out_kernel(const float* __restrict__ U,
                                                  const float* __restrict__ Wo,
                                                  const float* __restrict__ bo,
                                                  float* __restrict__ out) {
  const int pb = blockIdx.x;
  const float* ur = U + (size_t)pb * DM;
#pragma unroll
  for (int i = 0; i < 4; ++i) {
    const int o = threadIdx.x + i * 256;
    float acc = bo[o];
    for (int d = 0; d < DM; ++d)
      acc = fmaf(ur[d], Wo[(size_t)d * DM + o], acc);
    out[(size_t)pb * DM + o] = acc;
  }
}

// ---------------------------------------------------------------------------
extern "C" void kernel_launch(void* const* d_in, const int* in_sizes, int n_in,
                              void* d_out, int out_size, void* d_ws, size_t ws_size,
                              hipStream_t stream) {
  const float* q   = (const float*)d_in[0];
  const float* k   = (const float*)d_in[1];
  const float* v   = (const float*)d_in[2];
  const float* Wq  = (const float*)d_in[3];
  const float* bq  = (const float*)d_in[4];
  const float* Wk  = (const float*)d_in[5];
  const float* bk  = (const float*)d_in[6];
  const float* Wv  = (const float*)d_in[7];
  const float* bv  = (const float*)d_in[8];
  const float* Wo  = (const float*)d_in[9];
  const float* bo  = (const float*)d_in[10];
  const unsigned char* mask = (const unsigned char*)d_in[11];
  float* out = (float*)d_out;

  // Workspace carve-up (all 256B aligned)
  char* ws = (char*)d_ws;
  size_t off = 0;
  auto carve = [&](size_t bytes) { void* p = ws + off; off = (off + bytes + 255) & ~(size_t)255; return p; };
  float*  qh_ws     = (float*) carve((size_t)BB * DM * 4);            //  64 KB
  float*  sb_ws     = (float*) carve((size_t)BB * HH * 4);            //   1 KB
  bf16_t* qt_ws     = (bf16_t*)carve((size_t)BB * HH * DM * 2);       // 512 KB
  float*  scores_ws = (float*) carve((size_t)BB * HH * SS * 4);       //   4 MB
  bf16_t* wp_ws     = (bf16_t*)carve((size_t)BB * HH * SS * 2);       //   2 MB
  bf16_t* wm_ws     = (bf16_t*)carve((size_t)BB * HH * SS * 2);       //   2 MB
  float*  vbar_ws   = (float*) carve((size_t)2 * BB * HH * DM * 4);   //   2 MB
  float*  U_ws      = (float*) carve((size_t)2 * BB * DM * 4);        // 128 KB
  (void)ws_size; (void)in_sizes; (void)n_in; (void)out_size;

  qh_kernel     <<<dim3(BB),          256, 0, stream>>>(q, Wq, bq, qh_ws);
  qt_kernel     <<<dim3(HH, BB),      256, 0, stream>>>(qh_ws, Wk, bk, qt_ws, sb_ws);
  score_kernel  <<<dim3(SS/16/8, BB), 256, 0, stream>>>(k, qt_ws, sb_ws, scores_ws);
  softmax_kernel<<<dim3(BB * HH),     256, 0, stream>>>(scores_ws, mask, wp_ws, wm_ws);
  vbar_kernel   <<<dim3(DM/16/8, BB), 256, 0, stream>>>(v, wp_ws, wm_ws, vbar_ws);
  uproj_kernel  <<<dim3(2 * BB),      256, 0, stream>>>(vbar_ws, Wv, bv, U_ws);
  out_kernel    <<<dim3(2 * BB),      256, 0, stream>>>(U_ws, Wo, bo, out);
}
